// GridRender_31997506355428
// MI455X (gfx1250) — matrix operand order repaired and checked by reading, not compiled, not run on hardware
//
#include <hip/hip_runtime.h>
#include <hip/hip_bf16.h>

typedef __attribute__((ext_vector_type(16))) _Float16 v16h;
typedef __attribute__((ext_vector_type(8)))  _Float16 v8h;
typedef __attribute__((ext_vector_type(8)))  float    v8f;

#define GRID_NUM 16
#define G_TOTAL  4096
#define NS       32
#define CCH      64
#define NPTS     256
#define BATCH    2
#define GC       128   // g-columns handled per workgroup
#define AST      40    // padded LDS row stride (f16) for A  (rows=c, cols=n)
#define WST      40    // padded LDS row stride (f16) for Wt (rows=g, cols=n)

__global__ __launch_bounds__(256)
void gridrender_wmma(const float* __restrict__ feats,
                     const float* __restrict__ locs,
                     float* __restrict__ out)
{
    __shared__ __align__(16) _Float16 A_s[CCH * AST];   // A[c][n], f16
    __shared__ __align__(16) _Float16 W_s[GC * WST];    // W[g][n], f16 (transposed)
    __shared__ float locs_s[3 * NS];
    __shared__ float sum_s[2 * GC];
    __shared__ float recip_s[GC];

    const int tid   = threadIdx.x;
    const int t     = blockIdx.y;            // 0..511
    const int gbase = blockIdx.x * GC;       // chunk of 128 grid cells
    const int b     = t / NPTS;
    const int p     = t % NPTS;

    // ---- Phase 1: point coordinates (3 x 32 floats) ----
    if (tid < 3 * NS) {
        int d = tid / NS, n = tid % NS;
        locs_s[d * NS + n] = locs[(((size_t)b * 3 + d) * NPTS + p) * NS + n];
    }
    // ---- Phase 2: feats tile (64 x 32) -> f16 in LDS ----
    for (int e = tid; e < CCH * NS; e += 256) {
        int c = e >> 5, n = e & 31;
        A_s[c * AST + n] =
            (_Float16)feats[(((size_t)b * CCH + c) * NPTS + p) * NS + n];
    }
    __syncthreads();

    // ---- Phase 3: unnormalized tent weights w[n][g] for this g-chunk ----
    {
        const int gcol = tid & (GC - 1);
        const int half = tid >> 7;            // n in [0,16) or [16,32)
        const int g    = gbase + gcol;
        const float step = 2.0f / (float)(GRID_NUM - 1);
        const float gx = -1.0f + step * (float)(g >> 8);
        const float gy = -1.0f + step * (float)((g >> 4) & 15);
        const float gz = -1.0f + step * (float)(g & 15);
        const float invR = 8.66025403784438646763f;  // (GRID_NUM-1)/sqrt(3)
        float psum = 0.0f;
        #pragma unroll
        for (int j = 0; j < 16; ++j) {
            int n = half * 16 + j;
            float dx = locs_s[n]          - gx;
            float dy = locs_s[NS + n]     - gy;
            float dz = locs_s[2 * NS + n] - gz;
            float dist = sqrtf(dx * dx + dy * dy + dz * dz);
            float w = 1.0f - dist * invR;
            w = w > 0.0f ? w : 0.0f;
            psum += w;
            W_s[gcol * WST + n] = (_Float16)w;
        }
        sum_s[half * GC + gcol] = psum;
    }
    __syncthreads();
    if (tid < GC)
        recip_s[tid] = 1.0f / (sum_s[tid] + sum_s[GC + tid] + 1e-6f);
    __syncthreads();

    // ---- Phase 4: WMMA GEMM  out[c,g] = sum_n A[c,n] * W[n,g] ----
    const int wave   = tid >> 5;   // 0..7 -> N-tile of 16 g-columns
    const int lane   = tid & 31;
    const int lane_n = lane & 15;
    const bool hig   = lane >= 16;

    // B fragment (32x16): lane holds col N=lane%16, halves = K (hig?16:0)+0..15
    v16h bfrag;
    {
        const int grow = wave * 16 + lane_n;
        const int kb   = hig ? 16 : 0;
        v8h blo = *(const v8h*)&W_s[grow * WST + kb];
        v8h bhi = *(const v8h*)&W_s[grow * WST + kb + 8];
        #pragma unroll
        for (int i = 0; i < 8; ++i) { bfrag[i] = blo[i]; bfrag[8 + i] = bhi[i]; }
    }

    v8f acc[4];
    #pragma unroll
    for (int m = 0; m < 4; ++m)
        #pragma unroll
        for (int r = 0; r < 8; ++r) acc[m][r] = 0.0f;

    // A fragment (16x32): lane holds row M=lane%16,
    // halves 0..7 = K (hig?16:0)+0..7, halves 8..15 = K (hig?24:8)+0..7
    #pragma unroll
    for (int m = 0; m < 4; ++m) {
        const int row = m * 16 + lane_n;
        const int kb0 = hig ? 16 : 0;
        const int kb1 = hig ? 24 : 8;
        v8h alo = *(const v8h*)&A_s[row * AST + kb0];
        v8h ahi = *(const v8h*)&A_s[row * AST + kb1];
        v16h afrag;
        #pragma unroll
        for (int i = 0; i < 8; ++i) { afrag[i] = alo[i]; afrag[8 + i] = ahi[i]; }
        acc[m] = __builtin_amdgcn_wmma_f32_16x16x32_f16(
            false, afrag, false, bfrag, (short)0, acc[m], false, false);
    }

    // ---- Phase 5: fold in 1/sum normalization, stream out (non-temporal) ----
    const int gcol_out = wave * 16 + lane_n;
    const float rs   = recip_s[gcol_out];
    const int   gout = gbase + gcol_out;
    const int   moff = hig ? 8 : 0;
    #pragma unroll
    for (int m = 0; m < 4; ++m) {
        #pragma unroll
        for (int r = 0; r < 8; ++r) {
            int c = m * 16 + moff + r;
            float* dst = &out[((size_t)t * CCH + c) * (size_t)G_TOTAL + gout];
            __builtin_nontemporal_store(acc[m][r] * rs, dst);
        }
    }
}

extern "C" void kernel_launch(void* const* d_in, const int* in_sizes, int n_in,
                              void* d_out, int out_size, void* d_ws, size_t ws_size,
                              hipStream_t stream) {
    (void)in_sizes; (void)n_in; (void)out_size; (void)d_ws; (void)ws_size;
    const float* feats = (const float*)d_in[0];   // (B=2, C=64, NP=256, NS=32) f32
    const float* locs  = (const float*)d_in[1];   // (B=2, 3,    NP=256, NS=32) f32
    float* out = (float*)d_out;                   // (T=512, C=64, G=4096) f32
    dim3 grid(G_TOTAL / GC, BATCH * NPTS);        // (32, 512)
    gridrender_wmma<<<grid, dim3(256), 0, stream>>>(feats, locs, out);
}
